// PyramidVoxelConvolution_22093311771085
// MI455X (gfx1250) — matrix operand order repaired and checked by reading, not compiled, not run on hardware
//
#include <hip/hip_runtime.h>
#include <hip/hip_bf16.h>

// ---------------------------------------------------------------------------
// MI455X (gfx1250) e3nn-style equivariant 3D conv.
// Compute-bound (~3700 FLOP/byte, 453 GFLOP): all MACs go through
// v_wmma_f32_16x16x32_bf16. Input is re-laid-out into a zero-padded
// [52][52][52][128] bf16 volume (36MB, L2-resident on the 192MB L2) so each
// (dz,dy) halo row is one contiguous 13,312B block -> staged into LDS by the
// Tensor Data Mover (tensor_load_to_lds), double-buffered and overlapped with
// the WMMA stream. 8 wave32 per WGP workgroup, one 16-out-channel block each.
// ---------------------------------------------------------------------------

typedef __attribute__((ext_vector_type(16))) __bf16 v16bf;
typedef __attribute__((ext_vector_type(8)))  float  v8f;
typedef unsigned int u32x4 __attribute__((ext_vector_type(4)));
typedef int          i32x4 __attribute__((ext_vector_type(4)));
typedef int          i32x8 __attribute__((ext_vector_type(8)));

#define NVOX   110592                  // 48^3
#define CH     128
#define PD     52                      // padded extent (48 + 2*2 halo)
#define PVOX   (PD * PD * PD)          // 140,608 padded voxels
#define ROWB   (PD * CH * 2)           // 13,312 B per staged halo row
#define ROWQW  (ROWB / 8)              // 1664 qwords (TDM data_size=8B)
#define KFRAG  512                     // 16x32 bf16 elements per A fragment
#define KELEMS (125 * 8 * 4 * KFRAG)   // 2,048,000 bf16

#if __has_builtin(__builtin_amdgcn_tensor_load_to_lds)
#define USE_TDM 1
#else
#define USE_TDM 0
#endif

// ---------------------------------------------------------------------------
// Zero the padded input volume (borders must read as conv zero-padding).
// ---------------------------------------------------------------------------
__global__ __launch_bounds__(256) void zero_pad(uint4* __restrict__ p, int n) {
  int i = blockIdx.x * 256 + threadIdx.x;
  if (i < n) p[i] = make_uint4(0u, 0u, 0u, 0u);
}

// ---------------------------------------------------------------------------
// Phase 0: x [128][48^3] f32 -> xtp padded [52][52][52][128] bf16
// (channel-fastest). LDS-tiled so both global streams are coalesced.
// ---------------------------------------------------------------------------
__global__ __launch_bounds__(256) void transpose_cast(
    const float* __restrict__ x, __hip_bfloat16* __restrict__ xtp) {
  __shared__ __hip_bfloat16 tile[128 * 130];   // padded stride: no bank conflicts
  const int tid = threadIdx.x;
  const int vb  = blockIdx.x * 128;            // 864 blocks * 128 voxels

  for (int c0 = 0; c0 < 128; c0 += 2) {        // coalesced 512B global reads
    int c = c0 + (tid >> 7);
    int v = tid & 127;
    float f = x[(size_t)c * NVOX + vb + v];
    tile[v * 130 + c] = __float2bfloat16(f);
  }
  __syncthreads();
  for (int v0 = 0; v0 < 128; v0 += 4) {        // 64 thr * 1 dword = 256B rows
    int v  = v0 + (tid >> 6);
    int cp = tid & 63;
    int gv = vb + v;
    int zz = gv / 2304, yy = (gv / 48) % 48, xx = gv % 48;
    size_t po = ((size_t)((zz + 2) * PD + (yy + 2)) * PD + (xx + 2)) * CH;
    *(unsigned int*)((char*)xtp + (po + cp * 2) * 2) =
        *(const unsigned int*)((const char*)tile + (v * 130 + cp * 2) * 2);
  }
}

// ---------------------------------------------------------------------------
// Phase 1: build 125 x 128(in) x 128(out) conv kernel directly in WMMA
// A-fragment layout: kA[tap][outBlk(8)][kStep(4)][lane(32)][elem(16)] bf16.
// A 16x32 bf16 layout used here: lane = M + 16*khalf;
//   K = 16*khalf + e (contiguous k order, matches B fragment reads)
// ---------------------------------------------------------------------------
__global__ __launch_bounds__(256) void build_kernel(
    const float* __restrict__ w, const float* __restrict__ sc0,
    const float* __restrict__ sc1, __hip_bfloat16* __restrict__ kA) {
  int id = blockIdx.x * 256 + threadIdx.x;
  if (id >= KELEMS) return;

  const int e     = id & 15;
  const int lane  = (id >> 4) & 31;
  const int ks    = (id >> 9) & 3;
  const int ob    = (id >> 11) & 7;
  const int t     = id >> 14;
  const int khalf = lane >> 4;
  const int K     = 16 * khalf + e;
  const int o     = ob * 16 + (lane & 15);
  const int i     = ks * 32 + K;

  const int kd = t / 25, kh = (t / 5) % 5, kw = t % 5;
  const float fx = (float)(kd - 2), fy = (float)(kh - 2), fz = (float)(kw - 2);
  const float d  = sqrtf(fx * fx + fy * fy + fz * fz);
  const float nvox = 81.0f;                    // #{|v| <= 2.5} in [-2,2]^3

  const float step = 2.5f / 9.0f;
  const float Cemb = 1.14136f * __expf(2.0f);
  float emb[8];
#pragma unroll
  for (int j = 0; j < 8; ++j) {
    float diff = (d - step * (float)(j + 1)) / step;
    float t1 = diff + 1.0f, t2 = 1.0f - diff;
    float f1 = (t1 > 0.0f) ? __expf(-1.0f / t1) : 0.0f;
    float f2 = (t2 > 0.0f) ? __expf(-1.0f / t2) : 0.0f;
    emb[j] = Cemb * f1 * f2;
  }
  auto dotw = [&](int path, int u, int wc) -> float {
    int base = path * 1024 + u * 32 + wc;
    float s = 0.0f;
#pragma unroll
    for (int j = 0; j < 8; ++j) s += emb[j] * w[j * 4096 + base];
    return s / nvox;
  };

  const float rinv = (d > 0.0f) ? (1.0f / d) : 0.0f;
  const float unit[3] = { fx * rinv, fy * rinv, fz * rinv };

  float val;
  if (o < 32 && i < 32) {
    val = 0.125f * dotw(0, i, o);
  } else if (o >= 32 && i < 32) {
    int wc = (o - 32) / 3, n = (o - 32) % 3;
    val = 0.21650635f * unit[n] * dotw(1, i, wc);     // sqrt(3)/8
  } else if (o < 32) {
    int u = (i - 32) / 3, m = (i - 32) % 3;
    val = 0.125f * unit[m] * dotw(3, u, o);
  } else {
    int u = (i - 32) / 3, m = (i - 32) % 3;
    int wc = (o - 32) / 3, n = (o - 32) % 3;
    val = (m == n) ? 0.125f * dotw(2, u, wc) : 0.0f;
  }
  if (t == 62) {                               // center tap: self-connection
    const float inv = 0.1767766953f / nvox;    // 1/sqrt(32)/n_vox
    if (o < 32 && i < 32) {
      val += sc0[o * 32 + i] * inv;
    } else if (o >= 32 && i >= 32) {
      int uo = (o - 32) / 3, n = (o - 32) % 3;
      int ui = (i - 32) / 3, m = (i - 32) % 3;
      if (m == n) val += sc1[uo * 32 + ui] * inv;
    }
  }
  kA[id] = __float2bfloat16(val);
}

// ---------------------------------------------------------------------------
// TDM: one 1-D descriptor copies a contiguous 13,312B halo row -> LDS.
// D# per ISA 8.3/8.4: g0 = {count=1 | lds_addr | global_addr | type=2},
// g1 = {data_size=8B, tensor_dim0=tile_dim0=1664 qwords, stride=1664}.
// ---------------------------------------------------------------------------
#if USE_TDM
__device__ __forceinline__ void tdm_load_row(unsigned lds_off, const void* g) {
  unsigned long long ga = (unsigned long long)(size_t)g;
  u32x4 g0;
  g0[0] = 1u;                                    // count=1, user descriptor
  g0[1] = lds_off;                               // lds_addr (bytes)
  g0[2] = (unsigned)ga;                          // global_addr[31:0]
  g0[3] = (unsigned)(ga >> 32) | (2u << 30);     // global_addr[56:32] | type=2
  i32x8 g1 = { 0, 0, 0, 0, 0, 0, 0, 0 };
  g1[0] = (3 << 16);                             // data_size = 8B
  g1[1] = (int)((unsigned)ROWQW << 16);          // tensor_dim0[15:0]
  g1[3] = (int)((unsigned)ROWQW << 16);          // tile_dim0
  g1[5] = ROWQW;                                 // tensor_dim0_stride
  i32x4 z4 = { 0, 0, 0, 0 };
#if __has_include(<hip/amd_detail/amd_gfx1250_TDM.h>)
  i32x8 z8 = { 0, 0, 0, 0, 0, 0, 0, 0 };
  __builtin_amdgcn_tensor_load_to_lds(g0, g1, z4, z4, z8, 0);   // 6-arg toolchain
#else
  __builtin_amdgcn_tensor_load_to_lds(g0, g1, z4, z4, 0);       // 5-arg toolchain
#endif
}
#endif

// ---------------------------------------------------------------------------
// Phase 2: implicit-GEMM conv. WG = 256 thr (8 wave32) per output row (z,y):
// 48 voxels x all 128 out channels. 25 (dz,dy) stages, TDM double-buffered.
// ---------------------------------------------------------------------------
__global__ __launch_bounds__(256, 2) void conv_wmma(
    const __hip_bfloat16* __restrict__ xtp,  // padded [52][52][52][128] bf16
    const __hip_bfloat16* __restrict__ kA,   // [125][8][4][32][16] bf16
    float* __restrict__ out) {               // [128][48^3] f32
  __shared__ __align__(16) __hip_bfloat16 stage[2][PD * CH];   // 26,624 B

  const int tid  = threadIdx.x;
  const int wave = tid >> 5;                 // out-channel block 0..7
  const int lane = tid & 31;
  const int row  = blockIdx.x;               // 0..2303
  const int z = row / 48, y = row % 48;

  const int nloc = lane & 15;                // N within 16-voxel subtile
  // per-lane LDS base for B fragments: 16 consecutive in-channels
  const __hip_bfloat16* bl = &stage[0][0] + nloc * CH + ((lane >> 4) << 4);

  // (dz,dy) stage it -> contiguous padded row start (x spans padded 0..51)
  auto srcoff = [&](int it) -> size_t {
    return ((size_t)((z + it / 5) * PD + (y + it % 5)) * PD) * CH;
  };

  v8f acc0 = {}, acc1 = {}, acc2 = {};

#if USE_TDM
  const unsigned lds0 = (unsigned)(size_t)&stage[0][0];
  if (wave == 0) tdm_load_row(lds0, xtp + srcoff(0));          // prologue
#endif

  for (int it = 0; it < 25; ++it) {
    const int buf = it & 1;
#if USE_TDM
    if (wave == 0) __builtin_amdgcn_s_wait_tensorcnt(0);       // row `it` in LDS
    __syncthreads();            // publish row `it`; everyone done with buf^1
    if (wave == 0 && it + 1 < 25)
      tdm_load_row(lds0 + (unsigned)((buf ^ 1) * ROWB), xtp + srcoff(it + 1));
#else
    {   // manual fallback staging: contiguous copy, no bounds checks
      const uint4* src = (const uint4*)(xtp + srcoff(it));
      uint4* dst = (uint4*)&stage[buf][0];
      for (int c = tid; c < ROWB / 16; c += 256) dst[c] = src[c];
    }
    __syncthreads();
#endif

    const int tbase = it * 5;                 // tap = it*5 + dxi
#pragma unroll
    for (int dxi = 0; dxi < 5; ++dxi) {
      const __hip_bfloat16* ka =
          kA + (size_t)(((tbase + dxi) * 8 + wave) * 4) * KFRAG;
#pragma unroll
      for (int ks = 0; ks < 4; ++ks) {
        v16bf a = *(const v16bf*)(ka + ks * KFRAG + lane * 16);  // 32B/lane
        const __hip_bfloat16* bb = bl + buf * (PD * CH) + dxi * CH + ks * 32;
        v16bf b0 = *(const v16bf*)(bb);
        v16bf b1 = *(const v16bf*)(bb + 16 * CH);
        v16bf b2 = *(const v16bf*)(bb + 32 * CH);
        acc0 = __builtin_amdgcn_wmma_f32_16x16x32_bf16(
            false, a, false, b0, (short)0, acc0, false, false);
        acc1 = __builtin_amdgcn_wmma_f32_16x16x32_bf16(
            false, a, false, b1, (short)0, acc1, false, false);
        acc2 = __builtin_amdgcn_wmma_f32_16x16x32_bf16(
            false, a, false, b2, (short)0, acc2, false, false);
      }
    }
#if !USE_TDM
    __syncthreads();            // done reading buf before next overwrite
#endif
  }

  // C/D layout: VGPR r, lanes 0-15 -> M=r, lanes 16-31 -> M=8+r; N = lane&15
  const int mhi = (lane >> 4) * 8;
  const size_t rowoff = (size_t)row * 48;
#pragma unroll
  for (int r = 0; r < 8; ++r) {
    const int o = wave * 16 + mhi + r;
    float* op = out + (size_t)o * NVOX + rowoff + nloc;
    op[0]  = acc0[r];
    op[16] = acc1[r];
    op[32] = acc2[r];
  }
}

// ---------------------------------------------------------------------------
extern "C" void kernel_launch(void* const* d_in, const int* in_sizes, int n_in,
                              void* d_out, int out_size, void* d_ws, size_t ws_size,
                              hipStream_t stream) {
  (void)in_sizes; (void)n_in; (void)out_size; (void)ws_size;
  const float* x      = (const float*)d_in[0];   // [1,128,48,48,48]
  const float* weight = (const float*)d_in[1];   // [8,4096]
  const float* w_sc0  = (const float*)d_in[2];   // [32,32]
  const float* w_sc1  = (const float*)d_in[3];   // [32,32]
  float* out = (float*)d_out;

  __hip_bfloat16* xtp = (__hip_bfloat16*)d_ws;                 // 36.0 MB padded
  __hip_bfloat16* kA  =
      (__hip_bfloat16*)((char*)d_ws + (size_t)PVOX * CH * 2);  // +4.1 MB

  const int nq = (int)((size_t)PVOX * CH * 2 / 16);            // uint4 count
  zero_pad<<<(nq + 255) / 256, 256, 0, stream>>>((uint4*)xtp, nq);
  transpose_cast<<<NVOX / 128, 256, 0, stream>>>(x, xtp);
  build_kernel<<<(KELEMS + 255) / 256, 256, 0, stream>>>(weight, w_sc0, w_sc1, kA);
  conv_wmma<<<48 * 48, 256, 0, stream>>>(xtp, kA, out);
}
